// MultiHeadLatentAttention_76192719831539
// MI455X (gfx1250) — matrix operand order repaired and checked by reading, not compiled, not run on hardware
//
#include <hip/hip_runtime.h>
#include <hip/hip_bf16.h>
#include <cstdint>

typedef __bf16 bf16;
typedef __attribute__((ext_vector_type(16))) __bf16 bf16x16;
typedef __attribute__((ext_vector_type(8)))  float  f32x8;

#define B_DIM 2
#define T_DIM 2048
#define C_DIM 2048
#define NH 16
#define HD 128
#define KV_LORA 1024
#define KV_PE 256
#define KV_CONTENT 768
#define M_DIM (B_DIM * T_DIM) /* 4096 */

// ---------------------------------------------------------------------------
// CDNA5 async global->LDS copy (ASYNCcnt-tracked, no VGPR round trip).
// Per-lane: LDS[lds_off] = MEM[gaddr], 16 bytes. Generic LDS pointers carry the
// LDS byte offset in their low 32 bits (aperture in the high bits), so a
// truncating cast yields the VDST LDS address.
// ---------------------------------------------------------------------------
static __device__ __forceinline__ void async_ld_b128(unsigned lds_off,
                                                     const void* gaddr) {
  asm volatile("global_load_async_to_lds_b128 %0, %1, off"
               :: "v"(lds_off), "v"(gaddr) : "memory");
}
static __device__ __forceinline__ void wait_async0() {
  asm volatile("s_wait_asynccnt 0x0" ::: "memory");
}

// ---------------------------------------------------------------------------
// WMMA helpers (CDNA5 bf16 16x16x32, f32 accumulate)
// ---------------------------------------------------------------------------
static __device__ __forceinline__ f32x8 wmma_bf16f32(bf16x16 a, bf16x16 b, f32x8 c) {
  return __builtin_amdgcn_wmma_f32_16x16x32_bf16(false, a, false, b, (short)0, c,
                                                 false, false);
}

// A-fragment: 16(M) x 32(K) bf16 tile stored row-major in LDS with leading dim ldk.
// ISA layout: lanes 0-15 -> row M=lane, K 0..7 (V0-3) then 16..23 (V4-7);
//             lanes 16-31 -> K 8..15 then 24..31.
static __device__ __forceinline__ bf16x16 load_fragA(const bf16* t, int ldk,
                                                     int row0, int col0) {
  const int lane = threadIdx.x & 31;
  const int hi = lane >> 4, ln = lane & 15;
  const bf16* p = t + (size_t)(row0 + ln) * ldk + col0 + hi * 8;
  union { bf16x16 v; uint4 u[2]; } f;
  f.u[0] = *(const uint4*)p;
  f.u[1] = *(const uint4*)(p + 16);
  return f.v;
}

// B-fragment: 32(K) x 16(N) operand, stored transposed in LDS as [N][K] row-major.
// ISA layout: lanes 0-15 -> col N=lane, K 0..7 then 16..23; lanes 16-31 -> K 8..15 then 24..31.
static __device__ __forceinline__ bf16x16 load_fragB(const bf16* t, int ldk,
                                                     int row0, int col0) {
  const int lane = threadIdx.x & 31;
  const int hi = lane >> 4, ln = lane & 15;
  const bf16* p = t + (size_t)(row0 + ln) * ldk + col0 + hi * 8;
  union { bf16x16 v; uint4 u[2]; } f;
  f.u[0] = *(const uint4*)p;
  f.u[1] = *(const uint4*)(p + 16);
  return f.v;
}

// ---------------------------------------------------------------------------
// fp32 -> bf16 converts (plain + transposing, for the [N][K] weight layout)
// ---------------------------------------------------------------------------
__global__ void cvt_bf16_kernel(const float* __restrict__ in, bf16* __restrict__ out,
                                int n) {
  int i = blockIdx.x * 256 + threadIdx.x;
  if (i < n) out[i] = (bf16)in[i];
}

__global__ void cvt_t_bf16_kernel(const float* __restrict__ in, bf16* __restrict__ out,
                                  int R, int Ccol) {
  int i = blockIdx.x * 256 + threadIdx.x;
  if (i < R * Ccol) {
    int r = i / Ccol, c = i % Ccol;
    out[(size_t)c * R + r] = (bf16)in[i];
  }
}

// ---------------------------------------------------------------------------
// bf16 GEMM: C[M x N] = A[M x K] * Bt[N x K]^T.
// Workgroup tile 256x128, BK=32, double-buffered LDS filled by async copies.
// 8 waves in 4x2; each wave owns 64x64 = 4x4 WMMA tiles (16 fp32 accums),
// so each 32-K step is 8 LDS fragment loads : 16 WMMAs.
// OUT_MODE: 0 = fp32 row-major, 1 = bf16 row-major, 2 = bf16 transposed (ldc = M)
// ---------------------------------------------------------------------------
template <int OUT_MODE>
__global__ __launch_bounds__(256, 1)
void gemm_bf16_kernel(const bf16* __restrict__ A, int lda,
                      const bf16* __restrict__ Bt,
                      void* __restrict__ Cout, int ldc, int K) {
  __shared__ __align__(16) bf16 As[2][256 * 32];
  __shared__ __align__(16) bf16 Bs[2][128 * 32];
  const int tid = threadIdx.x;
  const int lane = tid & 31, w = tid >> 5;
  const int wr = w & 3, wc = w >> 2;
  const int hi = lane >> 4, ln = lane & 15;
  const int m0 = blockIdx.y * 256, n0 = blockIdx.x * 128;

  f32x8 acc[4][4];
#pragma unroll
  for (int r = 0; r < 4; ++r)
#pragma unroll
    for (int c = 0; c < 4; ++c) acc[r][c] = (f32x8){};

  // async-stage one 256x32 A slice + 128x32 B slice (b128 = 8 bf16 per lane)
  auto stage = [&](int buf, int k0) {
#pragma unroll
    for (int j = 0; j < 4; ++j) {            // A: 1024 chunks / 256 threads
      const int c = tid + 256 * j;
      const int row = c >> 2, col = (c & 3) * 8;
      async_ld_b128((unsigned)(uintptr_t)&As[buf][row * 32 + col],
                    A + (size_t)(m0 + row) * lda + k0 + col);
    }
#pragma unroll
    for (int j = 0; j < 2; ++j) {            // B: 512 chunks / 256 threads
      const int c = tid + 256 * j;
      const int row = c >> 2, col = (c & 3) * 8;
      async_ld_b128((unsigned)(uintptr_t)&Bs[buf][row * 32 + col],
                    Bt + (size_t)(n0 + row) * K + k0 + col);
    }
  };

  stage(0, 0);
  const int nk = K / 32;
  for (int it = 0; it < nk; ++it) {
    const int cur = it & 1;
    wait_async0();        // my async fills of As/Bs[cur] landed
    __syncthreads();      // everyone's fills landed; prior reads of [cur^1] done
    if (it + 1 < nk) stage(cur ^ 1, (it + 1) * 32);

    bf16x16 afr[4];
#pragma unroll
    for (int r = 0; r < 4; ++r)
      afr[r] = load_fragA(&As[cur][0], 32, wr * 64 + r * 16, 0);
#pragma unroll
    for (int c = 0; c < 4; ++c) {
      bf16x16 b = load_fragB(&Bs[cur][0], 32, wc * 64 + c * 16, 0);
#pragma unroll
      for (int r = 0; r < 4; ++r)
        acc[r][c] = wmma_bf16f32(afr[r], b, acc[r][c]);
    }
  }

#pragma unroll
  for (int r = 0; r < 4; ++r)
#pragma unroll
    for (int c = 0; c < 4; ++c)
#pragma unroll
      for (int i = 0; i < 8; ++i) {
        const int row = m0 + wr * 64 + r * 16 + i + 8 * hi;
        const int col = n0 + wc * 64 + c * 16 + ln;
        const float v = acc[r][c][i];
        if (OUT_MODE == 0)
          ((float*)Cout)[(size_t)row * ldc + col] = v;
        else if (OUT_MODE == 1)
          ((bf16*)Cout)[(size_t)row * ldc + col] = (bf16)v;
        else
          ((bf16*)Cout)[(size_t)col * ldc + row] = (bf16)v;
      }
}

// ---------------------------------------------------------------------------
// q: in-place per-head RMS-norm.  k: k_content + rotary(k_pe), then RMS-norm,
// written into the k_content buffer.  One 128-thread block per (b*T+t, mode).
// ---------------------------------------------------------------------------
__global__ __launch_bounds__(128, 1)
void finalize_qk_kernel(bf16* __restrict__ q, bf16* __restrict__ kc,
                        const bf16* __restrict__ kpe,
                        const float* __restrict__ cosp,
                        const float* __restrict__ sinp) {
  __shared__ float buf[128];
  __shared__ float wsum[4];
  const int mode = blockIdx.y;      // 0 = q, 1 = k
  const int m = blockIdx.x;         // b*T + t
  const int t = m & (T_DIM - 1);
  const int d = threadIdx.x;        // 0..127
  const int dd = d & 63;
  const float c = cosp[t * 64 + dd];
  const float s = sinp[t * 64 + dd];

  for (int h = 0; h < NH; ++h) {
    const size_t base = (size_t)m * C_DIM + h * HD;
    float val = (mode == 0) ? (float)q[base + d] : (float)kpe[base + d];
    buf[d] = val;
    __syncthreads();
    if (mode == 1) {
      float rot = (d < 64) ? (buf[d] * c + buf[d + 64] * s)
                           : (-buf[dd] * s + buf[d] * c);
      val = (float)kc[base + d] + rot;
    }
    float sq = val * val;
#pragma unroll
    for (int off = 16; off; off >>= 1) sq += __shfl_xor(sq, off, 32);
    if ((d & 31) == 0) wsum[d >> 5] = sq;
    __syncthreads();
    const float tot = wsum[0] + wsum[1] + wsum[2] + wsum[3];
    const float r = rsqrtf(tot * (1.0f / 128.0f) + 1.1920929e-07f);
    const float outv = val * r;
    if (mode == 0) q[base + d] = (bf16)outv;
    else           kc[base + d] = (bf16)outv;
  }
}

// ---------------------------------------------------------------------------
// Flash attention, causal.  Grid: (T/128, B*NH).  Block 256 = 8 waves.
// Each wave owns 16 q-rows; K/V streamed in 32-key chunks via async-to-LDS.
// qn, kn: [M][C] bf16 (per head a 128-col slice).  vt: [C][M] bf16 (transposed).
// ---------------------------------------------------------------------------
__global__ __launch_bounds__(256, 1)
void attn_kernel(const bf16* __restrict__ qn, const bf16* __restrict__ kn,
                 const bf16* __restrict__ vt, bf16* __restrict__ y) {
  __shared__ __align__(16) bf16 Qs[128 * 128];
  __shared__ __align__(16) bf16 Ks[32 * 128];
  __shared__ __align__(16) bf16 Vs[128 * 32];
  __shared__ __align__(16) bf16 Ps[8 * 16 * 32];

  const int bh = blockIdx.y;
  const int b = bh >> 4, h = bh & 15;
  const int q0 = blockIdx.x * 128;
  const int tid = threadIdx.x, lane = tid & 31, w = tid >> 5;
  const int hi = lane >> 4, ln = lane & 15;
  const float scale = 0.08838834764831845f; // 1/sqrt(128)
  const float NEG_BIG = -1.0e30f;

  {  // stage the 128x128 Q tile once
    const int r = tid >> 1, c0 = (tid & 1) * 64;
    const size_t g = (size_t)(b * T_DIM + q0 + r) * C_DIM + h * HD + c0;
#pragma unroll
    for (int i = 0; i < 8; ++i)
      *(uint4*)&Qs[r * 128 + c0 + i * 8] = *(const uint4*)&qn[g + i * 8];
  }

  f32x8 o[8];
  float mrow[8], lrow[8];
#pragma unroll
  for (int c = 0; c < 8; ++c) o[c] = (f32x8){};
#pragma unroll
  for (int i = 0; i < 8; ++i) { mrow[i] = NEG_BIG; lrow[i] = 0.0f; }

  const int nkb = q0 / 32 + 4;  // causal bound for this 128-row q tile
  for (int kb = 0; kb < nkb; ++kb) {
    __syncthreads();            // everyone done reading Ks/Vs of prev chunk
    {  // async-stage K chunk [t][d] (512 b128) and V chunk [d][t] (512 b128)
#pragma unroll
      for (int j = 0; j < 2; ++j) {
        const int c = tid + 256 * j;
        {
          const int row = c >> 4, col = (c & 15) * 8;
          async_ld_b128((unsigned)(uintptr_t)&Ks[row * 128 + col],
                        kn + (size_t)(b * T_DIM + kb * 32 + row) * C_DIM +
                            h * HD + col);
        }
        {
          const int row = c >> 2, col = (c & 3) * 8;
          async_ld_b128((unsigned)(uintptr_t)&Vs[row * 32 + col],
                        vt + (size_t)(h * HD + row) * M_DIM + b * T_DIM +
                            kb * 32 + col);
        }
      }
    }
    wait_async0();
    __syncthreads();

    // S = Q K^T : 16 q-rows x 32 keys per wave, K-depth 128 in 4 WMMA steps
    f32x8 s0 = (f32x8){}, s1 = (f32x8){};
#pragma unroll
    for (int d0 = 0; d0 < 128; d0 += 32) {
      bf16x16 a  = load_fragA(Qs, 128, w * 16, d0);
      bf16x16 k0 = load_fragB(Ks, 128, 0, d0);
      bf16x16 k1 = load_fragB(Ks, 128, 16, d0);
      s0 = wmma_bf16f32(a, k0, s0);
      s1 = wmma_bf16f32(a, k1, s1);
    }

    // online softmax across the 32 keys of this chunk (C layout: row=i+8*hi,
    // cols spread across lanes 0..15 of each half -> 16-lane shfl reductions)
#pragma unroll
    for (int i = 0; i < 8; ++i) {
      const int qt = q0 + w * 16 + i + 8 * hi;
      const int kt0 = kb * 32 + ln, kt1 = kt0 + 16;
      const bool ok0 = kt0 <= qt, ok1 = kt1 <= qt;
      const float v0 = ok0 ? s0[i] * scale : NEG_BIG;
      const float v1 = ok1 ? s1[i] * scale : NEG_BIG;
      float mx = fmaxf(v0, v1);
      mx = fmaxf(mx, __shfl_xor(mx, 1, 32));
      mx = fmaxf(mx, __shfl_xor(mx, 2, 32));
      mx = fmaxf(mx, __shfl_xor(mx, 4, 32));
      mx = fmaxf(mx, __shfl_xor(mx, 8, 32));
      const float mnew = fmaxf(mrow[i], mx);
      const float p0 = ok0 ? __expf(v0 - mnew) : 0.0f;
      const float p1 = ok1 ? __expf(v1 - mnew) : 0.0f;
      float ps = p0 + p1;
      ps += __shfl_xor(ps, 1, 32);
      ps += __shfl_xor(ps, 2, 32);
      ps += __shfl_xor(ps, 4, 32);
      ps += __shfl_xor(ps, 8, 32);
      const float alpha = __expf(mrow[i] - mnew);
      lrow[i] = lrow[i] * alpha + ps;
      mrow[i] = mnew;
#pragma unroll
      for (int c = 0; c < 8; ++c) o[c][i] *= alpha;
      // stash P (bf16) in LDS in A-fragment geometry: [16 rows][32 cols]
      Ps[w * 512 + (i + 8 * hi) * 32 + ln]      = (bf16)p0;
      Ps[w * 512 + (i + 8 * hi) * 32 + ln + 16] = (bf16)p1;
    }

    // O += P V : one A frag (16x32), 8 column tiles of V
    bf16x16 pa = load_fragA(&Ps[w * 512], 32, 0, 0);
#pragma unroll
    for (int c = 0; c < 8; ++c) {
      bf16x16 bv = load_fragB(Vs, 32, c * 16, 0);
      o[c] = wmma_bf16f32(pa, bv, o[c]);
    }
  }

#pragma unroll
  for (int c = 0; c < 8; ++c)
#pragma unroll
    for (int i = 0; i < 8; ++i) {
      const float val = o[c][i] / lrow[i];
      const int row = b * T_DIM + q0 + w * 16 + i + 8 * hi;
      y[(size_t)row * C_DIM + h * HD + c * 16 + ln] = (bf16)val;
    }
}

// ---------------------------------------------------------------------------
// Host side
// ---------------------------------------------------------------------------
extern "C" void kernel_launch(void* const* d_in, const int* in_sizes, int n_in,
                              void* d_out, int out_size, void* d_ws, size_t ws_size,
                              hipStream_t stream) {
  (void)in_sizes; (void)n_in; (void)out_size; (void)ws_size;
  const float* x      = (const float*)d_in[0];
  const float* cosp   = (const float*)d_in[1];
  const float* sinp   = (const float*)d_in[2];
  const float* wq_a   = (const float*)d_in[3];
  const float* wq_b   = (const float*)d_in[4];
  const float* wkv_a  = (const float*)d_in[5];
  const float* wk_b   = (const float*)d_in[6];
  const float* wkpe_b = (const float*)d_in[7];
  const float* wv_b   = (const float*)d_in[8];
  const float* wo     = (const float*)d_in[9];

  char* ws = (char*)d_ws;
  size_t off = 0;
  auto alloc = [&](size_t elems) {
    bf16* p = (bf16*)(ws + off);
    off += (elems * sizeof(bf16) + 255) & ~(size_t)255;
    return p;
  };
  bf16* xb      = alloc((size_t)M_DIM * C_DIM);
  bf16* wq_aT   = alloc((size_t)KV_LORA * C_DIM);
  bf16* wq_bT   = alloc((size_t)C_DIM * KV_LORA);
  bf16* wkv_aT  = alloc((size_t)KV_LORA * C_DIM);
  bf16* wk_bT   = alloc((size_t)C_DIM * KV_CONTENT);
  bf16* wkpe_bT = alloc((size_t)C_DIM * KV_PE);
  bf16* wv_bT   = alloc((size_t)C_DIM * KV_CONTENT);
  bf16* woT     = alloc((size_t)C_DIM * C_DIM);
  bf16* qlat    = alloc((size_t)M_DIM * KV_LORA);
  bf16* kvb     = alloc((size_t)M_DIM * KV_LORA);
  bf16* qb      = alloc((size_t)M_DIM * C_DIM);
  bf16* kcb     = alloc((size_t)M_DIM * C_DIM);
  bf16* kpeb    = alloc((size_t)M_DIM * C_DIM);
  bf16* vtb     = alloc((size_t)C_DIM * M_DIM);
  bf16* yb      = alloc((size_t)M_DIM * C_DIM);

  const dim3 blk(256);
  auto g1 = [](int n) { return dim3((unsigned)((n + 255) / 256)); };

  // bf16 conversions (weights transposed to [N][K])
  cvt_bf16_kernel<<<g1(M_DIM * C_DIM), blk, 0, stream>>>(x, xb, M_DIM * C_DIM);
  cvt_t_bf16_kernel<<<g1(C_DIM * KV_LORA), blk, 0, stream>>>(wq_a, wq_aT, C_DIM, KV_LORA);
  cvt_t_bf16_kernel<<<g1(KV_LORA * C_DIM), blk, 0, stream>>>(wq_b, wq_bT, KV_LORA, C_DIM);
  cvt_t_bf16_kernel<<<g1(C_DIM * KV_LORA), blk, 0, stream>>>(wkv_a, wkv_aT, C_DIM, KV_LORA);
  cvt_t_bf16_kernel<<<g1(KV_CONTENT * C_DIM), blk, 0, stream>>>(wk_b, wk_bT, KV_CONTENT, C_DIM);
  cvt_t_bf16_kernel<<<g1(KV_PE * C_DIM), blk, 0, stream>>>(wkpe_b, wkpe_bT, KV_PE, C_DIM);
  cvt_t_bf16_kernel<<<g1(KV_CONTENT * C_DIM), blk, 0, stream>>>(wv_b, wv_bT, KV_CONTENT, C_DIM);
  cvt_t_bf16_kernel<<<g1(C_DIM * C_DIM), blk, 0, stream>>>(wo, woT, C_DIM, C_DIM);

  // projections (all M = 4096, workgroup tile 256x128)
  gemm_bf16_kernel<1><<<dim3(KV_LORA / 128, M_DIM / 256), blk, 0, stream>>>(
      xb, C_DIM, wq_aT, qlat, KV_LORA, C_DIM);                        // q_latent
  gemm_bf16_kernel<1><<<dim3(C_DIM / 128, M_DIM / 256), blk, 0, stream>>>(
      qlat, KV_LORA, wq_bT, qb, C_DIM, KV_LORA);                      // q
  gemm_bf16_kernel<1><<<dim3(KV_LORA / 128, M_DIM / 256), blk, 0, stream>>>(
      xb, C_DIM, wkv_aT, kvb, KV_LORA, C_DIM);                        // kv_latent
  gemm_bf16_kernel<1><<<dim3(C_DIM / 128, M_DIM / 256), blk, 0, stream>>>(
      kvb + KV_PE, KV_LORA, wk_bT, kcb, C_DIM, KV_CONTENT);           // k_content
  gemm_bf16_kernel<1><<<dim3(C_DIM / 128, M_DIM / 256), blk, 0, stream>>>(
      kvb, KV_LORA, wkpe_bT, kpeb, C_DIM, KV_PE);                     // k_pe
  gemm_bf16_kernel<2><<<dim3(C_DIM / 128, M_DIM / 256), blk, 0, stream>>>(
      kvb + KV_PE, KV_LORA, wv_bT, vtb, M_DIM, KV_CONTENT);           // v (transposed)

  // rotary + add + rms-norm
  finalize_qk_kernel<<<dim3(M_DIM, 2), dim3(128), 0, stream>>>(qb, kcb, kpeb,
                                                               cosp, sinp);

  // flash attention
  attn_kernel<<<dim3(T_DIM / 128, B_DIM * NH), blk, 0, stream>>>(qb, kcb, vtb, yb);

  // output projection -> fp32 d_out
  gemm_bf16_kernel<0><<<dim3(C_DIM / 128, M_DIM / 256), blk, 0, stream>>>(
      yb, C_DIM, woT, d_out, C_DIM, C_DIM);
}
